// GraphAttentionLayer_7885559955756
// MI455X (gfx1250) — matrix operand order repaired and checked by reading, not compiled, not run on hardware
//
#include <hip/hip_runtime.h>
#include <hip/hip_bf16.h>

// Problem constants (B,S,D,H,C from the reference)
#define BB 8
#define SS 1024
#define DD 768
#define HH 12
#define CC 64
#define NEG_SLOPE 0.2f

#define LDS_PITCH 40   // elements per N-row of staged tile (80B, 16B-aligned, conflict-skewed)
#define TILE_ELEMS (64 * LDS_PITCH)

typedef __attribute__((ext_vector_type(16))) __bf16 v16bf;
typedef __attribute__((ext_vector_type(8)))  float  v8f;
typedef __attribute__((ext_vector_type(4)))  int    int4v;

union Frag16 { unsigned short u[16]; unsigned int w[8]; int4v q[2]; v16bf v; };

__device__ __forceinline__ unsigned short f2bf(float f) {
  unsigned int x = __float_as_uint(f);
  x += 0x7FFFu + ((x >> 16) & 1u);            // round-to-nearest-even
  return (unsigned short)(x >> 16);
}

// ---------------- fp32 -> bf16 conversions ----------------
__global__ __launch_bounds__(256) void cvt_f32_bf16(const float* __restrict__ src,
                                                    unsigned short* __restrict__ dst, int n) {
  int i = blockIdx.x * 256 + threadIdx.x;
  if (i < n) dst[i] = f2bf(src[i]);
}

// strided destination (fills a column block of the concat buffer)
__global__ __launch_bounds__(256) void cvt_f32_bf16_cols(const float* __restrict__ src,
                                                         unsigned short* __restrict__ dst,
                                                         int rows, int cols, int dld, int coff) {
  int i = blockIdx.x * 256 + threadIdx.x;
  if (i < rows * cols) {
    int r = i / cols, c = i % cols;
    dst[(size_t)r * dld + coff + c] = f2bf(src[i]);
  }
}

// stage a 32(K) x 64(N) bf16 tile into LDS, transposed to N-major.
// 256 threads: thread t loads 8 contiguous bf16 of row k0+(t>>3) starting at col (t&7)*8.
__device__ __forceinline__ void stage_Btile(const unsigned short* __restrict__ Bsrc,
                                            int ldb, int k0, int kmax,
                                            unsigned short* __restrict__ lds, int t) {
  int kl = t >> 3;                 // 0..31
  int nl = (t & 7) * 8;            // 0..56
  const unsigned int* src = (const unsigned int*)(Bsrc + (size_t)(k0 + kl) * ldb);
  unsigned int d[4] = {src[nl / 2 + 0], src[nl / 2 + 1], src[nl / 2 + 2], src[nl / 2 + 3]};
  if (k0 + 32 < kmax)              // prefetch the K-step after next (global_prefetch_b8)
    __builtin_prefetch(Bsrc + (size_t)(k0 + 32 + kl) * ldb + nl, 0, 1);
#pragma unroll
  for (int e = 0; e < 8; ++e)
    lds[(nl + e) * LDS_PITCH + kl] = (unsigned short)(d[e >> 1] >> ((e & 1) * 16));
}

// read one 32x16 B fragment (n-tile nt) from the N-major LDS tile: 2x ds_load_b128
__device__ __forceinline__ Frag16 read_Bfrag(const unsigned short* __restrict__ lds,
                                             int nt, int l15, int half) {
  Frag16 f;
  const int4v* bp = (const int4v*)(lds + (nt * 16 + l15) * LDS_PITCH + half * 8);
  f.q[0] = bp[0];
  f.q[1] = bp[2];
  return f;
}

// ---------------- generic bf16 WMMA GEMM: C = A @ B (+bias) ----------------
// Block = 8 waves = 128(M) x 64(N) tile; double-buffered LDS B tile,
// one barrier per K-step; 4 back-to-back WMMAs per step.
__global__ __launch_bounds__(256) void gemm_bf16_wmma(
    const unsigned short* __restrict__ A, int lda,
    const unsigned short* __restrict__ Bm, int ldb,
    const float* __restrict__ bias,
    float* __restrict__ Cf, int ldcf,
    unsigned short* __restrict__ Cbf, int ldcbf, int coff,
    int M, int N, int K) {
  __shared__ unsigned short ldsB[2 * TILE_ELEMS];
  const int t    = threadIdx.x;
  const int lane = t & 31;
  const int wave = t >> 5;
  const int tilesN = N >> 6;
  const int bm = blockIdx.x / tilesN;
  const int bn = blockIdx.x % tilesN;
  const int half = lane >> 4;
  const int l15  = lane & 15;

  v8f acc[4];
#pragma unroll
  for (int nt = 0; nt < 4; ++nt) acc[nt] = (v8f){0,0,0,0,0,0,0,0};

  const unsigned short* Arow = A + (size_t)(bm * 128 + wave * 16 + l15) * lda;
  const unsigned short* Bcol = Bm + bn * 64;

  stage_Btile(Bcol, ldb, 0, K, ldsB, t);          // prologue: fill buffer 0
  __syncthreads();

  for (int k0 = 0; k0 < K; k0 += 32) {
    unsigned short* cur = ldsB + ((k0 >> 5) & 1) * TILE_ELEMS;
    unsigned short* nxt = ldsB + (((k0 >> 5) & 1) ^ 1) * TILE_ELEMS;
    if (k0 + 32 < K) stage_Btile(Bcol, ldb, k0 + 32, K, nxt, t);  // overlap w/ compute

    // A fragment: bf16 A layout — lanes 0-15 K{0..7,16..23}, lanes 16-31 K{8..15,24..31}
    Frag16 a;
#pragma unroll
    for (int v = 0; v < 8; ++v) {
      int kk = (v < 4) ? (2 * v + 8 * half) : (2 * v + 8 + 8 * half);
      a.w[v] = *(const unsigned int*)(Arow + k0 + kk);   // merges into 2x b128
    }
    // load all 4 B fragments first (one ds clause, one wait), then 4 WMMAs back-to-back
    Frag16 bf[4];
#pragma unroll
    for (int nt = 0; nt < 4; ++nt) bf[nt] = read_Bfrag(cur, nt, l15, half);
#pragma unroll
    for (int nt = 0; nt < 4; ++nt)
      acc[nt] = __builtin_amdgcn_wmma_f32_16x16x32_bf16(false, a.v, false, bf[nt].v,
                                                        (short)0, acc[nt], false, false);
    __syncthreads();   // next tile staged AND this tile's reads complete
  }
#pragma unroll
  for (int nt = 0; nt < 4; ++nt) {
    int col = bn * 64 + nt * 16 + l15;
    float bv = bias ? bias[col] : 0.f;
#pragma unroll
    for (int r = 0; r < 8; ++r) {
      int row = bm * 128 + wave * 16 + r + 8 * half;  // f32 D: M = r + 8*(lane>=16)
      float val = acc[nt][r] + bv;
      if (Cf)  Cf[(size_t)row * ldcf + col] = val;
      if (Cbf) Cbf[(size_t)row * ldcbf + coff + col] = f2bf(val);
    }
  }
}

// ---------------- a_src / a_dst: per (b,h,s) dot with attention vectors -------
__global__ __launch_bounds__(256) void gat_scores(
    const unsigned short* __restrict__ xbf,
    const float* __restrict__ att_src, const float* __restrict__ att_dst,
    float* __restrict__ a_src, float* __restrict__ a_dst) {
  int i = blockIdx.x * 256 + threadIdx.x;         // flat over B*H*S
  if (i >= BB * HH * SS) return;
  int s = i % SS, h = (i / SS) % HH, b = i / (SS * HH);
  const unsigned int* xr =
      (const unsigned int*)(xbf + ((size_t)b * SS + s) * DD + h * CC);
  const float* wsv = att_src + h * CC;
  const float* wdv = att_dst + h * CC;
  float as = 0.f, ad = 0.f;
#pragma unroll 8
  for (int c = 0; c < CC / 2; ++c) {
    unsigned int pk = xr[c];
    float x0 = __uint_as_float(pk << 16);
    float x1 = __uint_as_float(pk & 0xFFFF0000u);
    as += x0 * wsv[2 * c] + x1 * wsv[2 * c + 1];
    ad += x0 * wdv[2 * c] + x1 * wdv[2 * c + 1];
  }
  a_src[i] = as;
  a_dst[i] = ad;
}

// ---------------- softmax row stats: m_i = max_j lrelu(d_i + s_j), 1/l_i ------
__global__ __launch_bounds__(256) void softmax_stats(
    const float* __restrict__ a_src, const float* __restrict__ a_dst,
    float* __restrict__ m_out, float* __restrict__ linv_out) {
  int lane = threadIdx.x & 31;
  int wave = threadIdx.x >> 5;
  int row = blockIdx.x * 8 + wave;                // flat over B*H*S
  if (row >= BB * HH * SS) return;
  int bh = row / SS;
  const float* srow = a_src + (size_t)bh * SS;
  float d = a_dst[row];
  float mx = -3.4e38f;
  for (int j = lane; j < SS; j += 32) {
    float sc = d + srow[j];
    sc = sc > 0.f ? sc : NEG_SLOPE * sc;
    mx = fmaxf(mx, sc);
  }
#pragma unroll
  for (int o = 16; o > 0; o >>= 1) mx = fmaxf(mx, __shfl_xor(mx, o, 32));
  float sum = 0.f;
  for (int j = lane; j < SS; j += 32) {
    float sc = d + srow[j];
    sc = sc > 0.f ? sc : NEG_SLOPE * sc;
    sum += __expf(sc - mx);
  }
#pragma unroll
  for (int o = 16; o > 0; o >>= 1) sum += __shfl_xor(sum, o, 32);
  if (lane == 0) { m_out[row] = mx; linv_out[row] = 1.f / sum; }
}

// ---------------- attention aggregation: gat = softmax(P) @ x_head + b_gat ----
// Block = 8 waves = one (b,h) and a 128-row i-slab. Double-buffered LDS x tile;
// P A-fragment synthesized in registers (exp on VALU) overlaps LDS traffic.
__global__ __launch_bounds__(256) void gat_attn_aggregate(
    const unsigned short* __restrict__ xbf,
    const float* __restrict__ a_src, const float* __restrict__ a_dst,
    const float* __restrict__ m_arr, const float* __restrict__ linv,
    const float* __restrict__ b_gat,
    unsigned short* __restrict__ gat_bf) {
  __shared__ unsigned short ldsX[2 * TILE_ELEMS];
  const int t    = threadIdx.x;
  const int lane = t & 31;
  const int wave = t >> 5;
  const int ib = blockIdx.x % (SS / 128);         // 8 slabs of 128 rows
  const int h  = (blockIdx.x / (SS / 128)) % HH;
  const int b  = blockIdx.x / ((SS / 128) * HH);
  const int half = lane >> 4;
  const int l15  = lane & 15;

  const int i  = ib * 128 + wave * 16 + l15;      // A-matrix row for this lane
  const int bh = b * HH + h;
  const float di = a_dst[(size_t)bh * SS + i];
  const float mi = m_arr[(size_t)bh * SS + i];
  const float li = linv [(size_t)bh * SS + i];
  const float* srow = a_src + (size_t)bh * SS;
  const unsigned short* xh = xbf + (size_t)b * SS * DD + h * CC;

  v8f acc[4];
#pragma unroll
  for (int nt = 0; nt < 4; ++nt) acc[nt] = (v8f){0,0,0,0,0,0,0,0};

  stage_Btile(xh, DD, 0, SS, ldsX, t);            // prologue: fill buffer 0
  __syncthreads();

  for (int j0 = 0; j0 < SS; j0 += 32) {
    unsigned short* cur = ldsX + ((j0 >> 5) & 1) * TILE_ELEMS;
    unsigned short* nxt = ldsX + (((j0 >> 5) & 1) ^ 1) * TILE_ELEMS;
    if (j0 + 32 < SS) stage_Btile(xh, DD, j0 + 32, SS, nxt, t);

    Frag16 p;                                      // probabilities at A-layout positions
#pragma unroll
    for (int v = 0; v < 8; ++v) {
      int kk = (v < 4) ? (2 * v + 8 * half) : (2 * v + 8 + 8 * half);
      int j = j0 + kk;
      float s0 = di + srow[j];
      float s1 = di + srow[j + 1];
      s0 = s0 > 0.f ? s0 : NEG_SLOPE * s0;
      s1 = s1 > 0.f ? s1 : NEG_SLOPE * s1;
      p.u[2 * v]     = f2bf(__expf(s0 - mi) * li);
      p.u[2 * v + 1] = f2bf(__expf(s1 - mi) * li);
    }
    Frag16 xf[4];
#pragma unroll
    for (int nt = 0; nt < 4; ++nt) xf[nt] = read_Bfrag(cur, nt, l15, half);
#pragma unroll
    for (int nt = 0; nt < 4; ++nt)
      acc[nt] = __builtin_amdgcn_wmma_f32_16x16x32_bf16(false, p.v, false, xf[nt].v,
                                                        (short)0, acc[nt], false, false);
    __syncthreads();
  }
#pragma unroll
  for (int nt = 0; nt < 4; ++nt) {
    int ch = h * CC + nt * 16 + l15;
    float bv = b_gat[ch];
#pragma unroll
    for (int r = 0; r < 8; ++r) {
      int row = ib * 128 + wave * 16 + r + 8 * half;
      gat_bf[((size_t)b * SS + row) * DD + ch] = f2bf(acc[nt][r] + bv);
    }
  }
}

// -----------------------------------------------------------------------------
extern "C" void kernel_launch(void* const* d_in, const int* in_sizes, int n_in,
                              void* d_out, int out_size, void* d_ws, size_t ws_size,
                              hipStream_t stream) {
  const float* hs    = (const float*)d_in[0];   // [B,S,D]
  const float* to    = (const float*)d_in[1];   // [B,S,D]
  const float* Wgat  = (const float*)d_in[2];   // [D,D]
  const float* attS  = (const float*)d_in[3];   // [H,C]
  const float* attD  = (const float*)d_in[4];   // [H,C]
  const float* bgat  = (const float*)d_in[5];   // [D]
  const float* Wproj = (const float*)d_in[6];   // [D,D]
  const float* bproj = (const float*)d_in[7];   // [D]
  const float* Wfus  = (const float*)d_in[8];   // [2D,D]
  const float* bfus  = (const float*)d_in[9];   // [D]
  float* out = (float*)d_out;

  const int M = BB * SS;                         // 8192
  char* ws = (char*)d_ws;
  size_t off = 0;
  auto carve = [&](size_t bytes) -> char* {
    char* p = ws + off;
    off += (bytes + 255) & ~(size_t)255;
    return p;
  };
  unsigned short* hs_bf   = (unsigned short*)carve((size_t)M * DD * 2);
  unsigned short* Wgat_bf = (unsigned short*)carve((size_t)DD * DD * 2);
  unsigned short* Wproj_bf= (unsigned short*)carve((size_t)DD * DD * 2);
  unsigned short* Wfus_bf = (unsigned short*)carve((size_t)2 * DD * DD * 2);
  unsigned short* x_bf    = (unsigned short*)carve((size_t)M * DD * 2);
  float* a_src = (float*)carve((size_t)BB * HH * SS * 4);
  float* a_dst = (float*)carve((size_t)BB * HH * SS * 4);
  float* m_arr = (float*)carve((size_t)BB * HH * SS * 4);
  float* l_inv = (float*)carve((size_t)BB * HH * SS * 4);
  unsigned short* gat_bf  = (unsigned short*)carve((size_t)M * DD * 2);
  unsigned short* concat  = (unsigned short*)carve((size_t)M * 2 * DD * 2);
  (void)ws_size; (void)in_sizes; (void)n_in; (void)out_size;

  // conversions
  cvt_f32_bf16<<<(M * DD + 255) / 256, 256, 0, stream>>>(hs, hs_bf, M * DD);
  cvt_f32_bf16<<<(DD * DD + 255) / 256, 256, 0, stream>>>(Wgat, Wgat_bf, DD * DD);
  cvt_f32_bf16<<<(DD * DD + 255) / 256, 256, 0, stream>>>(Wproj, Wproj_bf, DD * DD);
  cvt_f32_bf16<<<(2 * DD * DD + 255) / 256, 256, 0, stream>>>(Wfus, Wfus_bf, 2 * DD * DD);
  cvt_f32_bf16_cols<<<(M * DD + 255) / 256, 256, 0, stream>>>(to, concat, M, DD, 2 * DD, 0);

  // GEMM1: x = hs @ W_gat  -> bf16 only
  gemm_bf16_wmma<<<(M / 128) * (DD / 64), 256, 0, stream>>>(
      hs_bf, DD, Wgat_bf, DD, nullptr,
      nullptr, 0, x_bf, DD, 0, M, DD, DD);

  // attention score coefficients + softmax stats
  gat_scores<<<(BB * HH * SS + 255) / 256, 256, 0, stream>>>(x_bf, attS, attD, a_src, a_dst);
  softmax_stats<<<(BB * HH * SS + 7) / 8, 256, 0, stream>>>(a_src, a_dst, m_arr, l_inv);

  // attention aggregation (WMMA): gat = softmax(lrelu(a_dst_i + a_src_j)) @ x + b_gat
  gat_attn_aggregate<<<BB * HH * (SS / 128), 256, 0, stream>>>(
      x_bf, a_src, a_dst, m_arr, l_inv, bgat, gat_bf);

  // GEMM2: graph_output = gat @ W_proj + b_proj  -> bf16 into concat[:, D:2D]
  gemm_bf16_wmma<<<(M / 128) * (DD / 64), 256, 0, stream>>>(
      gat_bf, DD, Wproj_bf, DD, bproj,
      nullptr, 0, concat, 2 * DD, DD, M, DD, DD);

  // GEMM3: fused = concat @ W_fus + b_fus -> f32 output
  gemm_bf16_wmma<<<(M / 128) * (DD / 64), 256, 0, stream>>>(
      concat, 2 * DD, Wfus_bf, DD, bfus,
      out, DD, nullptr, 0, 0, M, DD, 2 * DD);
}
